// CaptionHead_35811437314281
// MI455X (gfx1250) — matrix operand order repaired and checked by reading, not compiled, not run on hardware
//
#include <hip/hip_runtime.h>
#include <hip/hip_bf16.h>
#include <math.h>

// Problem constants (match the reference setup).
#define F_DIM   512
#define C_DIM   512
#define NCAP_N  2048

typedef __attribute__((ext_vector_type(16))) _Float16 v16h;
typedef __attribute__((ext_vector_type(8)))  _Float16 v8h;
typedef __attribute__((ext_vector_type(8)))  float    v8f;

// ---------------------------------------------------------------------------
// Kernel 1: gather voxel features to points, L2-normalize, convert to f16.
// One block (256 threads) per point row of 512 floats.
// ---------------------------------------------------------------------------
__global__ __launch_bounds__(256) void k_gather_norm(
    const float* __restrict__ feats, const int* __restrict__ v2p,
    _Float16* __restrict__ af, int P) {
  const int p = blockIdx.x;
  if (p >= P) return;
  const int v = v2p[p];
  const float* src = feats + (size_t)v * F_DIM;
  const int t = threadIdx.x;
  float x0 = src[t];
  float x1 = src[t + 256];
  float ss = x0 * x0 + x1 * x1;
  #pragma unroll
  for (int off = 16; off > 0; off >>= 1) ss += __shfl_xor(ss, off, 32);
  __shared__ float wsum[8];
  const int wave = t >> 5, lane = t & 31;
  if (lane == 0) wsum[wave] = ss;
  __syncthreads();
  float tot = 0.f;
  #pragma unroll
  for (int i = 0; i < 8; ++i) tot += wsum[i];
  const float inv = 1.0f / fmaxf(sqrtf(tot), 1e-12f);
  _Float16* dst = af + (size_t)p * F_DIM;
  dst[t]       = (_Float16)(x0 * inv);
  dst[t + 256] = (_Float16)(x1 * inv);
}

// ---------------------------------------------------------------------------
// Kernel 2: convert caption embeddings (already normalized) to f16.
// ---------------------------------------------------------------------------
__global__ __launch_bounds__(256) void k_ce_to_f16(
    const float* __restrict__ ce, _Float16* __restrict__ ce16, int n) {
  int i = blockIdx.x * 256 + threadIdx.x;
  if (i < n) ce16[i] = (_Float16)ce[i];
}

// ---------------------------------------------------------------------------
// Kernel 3: WMMA GEMM (af @ ce^T) * scale, fused row log_softmax.
// Block = 256 threads = 8 waves. Block owns 16 rows x all 512 cols.
// The 16x512 f16 A tile is staged in LDS once (all waves share it), so the
// inner loop is: 2x ds_load_b128 (A frag) + 8x global b128 (B stream, L2-hot)
// + 4x v_wmma_f32_16x16x32_f16.
// ---------------------------------------------------------------------------
__global__ __launch_bounds__(256) void k_gemm_logsoftmax(
    const _Float16* __restrict__ af, const _Float16* __restrict__ ce16,
    const float* __restrict__ logit_scale_log,
    float* __restrict__ scores, int P) {
  // +8 half pad -> row stride 1040 B: 16B-aligned and bank-conflict-free
  // for the 16-lane b128 fragment reads (lane l hits banks 4l..4l+3 mod 64).
  __shared__ __align__(16) _Float16 aT[16][F_DIM + 8];
  __shared__ float lds[16][C_DIM + 4];

  const int pt    = blockIdx.x;           // row tile (16 rows)
  const int tid   = threadIdx.x;
  const int wave  = tid >> 5;
  const int lane  = tid & 31;
  const int l16   = lane & 15;            // M (A) / N (B) / N (C,D)
  const int lhalf = lane >> 4;            // 0: lanes 0-15, 1: lanes 16-31

  const float scale = expf(logit_scale_log[0]);

  // ---- Stage the 16x512 A tile into LDS (16 KB), 64 B per thread chunk ----
  {
    const int row   = tid >> 4;           // 0..15
    const int chunk = (tid & 15) * 32;    // halves (64 B)
    const uint4* src =
        (const uint4*)(af + ((size_t)(pt * 16 + row)) * F_DIM + chunk);
    uint4* dst = (uint4*)&aT[row][chunk];
    dst[0] = src[0];
    dst[1] = src[1];
    dst[2] = src[2];
    dst[3] = src[3];
  }
  __syncthreads();

  v8f acc[4] = {};  // 4 column tiles, each 16x16 f32 accumulator

  for (int k0 = 0; k0 < F_DIM; k0 += 32) {
    // A fragment (CDNA5 16-bit A 16x32 layout): lane holds row M = l16,
    // K chunks [k0+kb8, +8) and [k0+16+kb8, +8), kb8 = lhalf*8.
    const _Float16* ap = &aT[l16][k0 + lhalf * 8];
    v8h a0 = *(const v8h*)(ap);
    v8h a1 = *(const v8h*)(ap + 16);
    v16h a = __builtin_shufflevector(a0, a1, 0, 1, 2, 3, 4, 5, 6, 7,
                                     8, 9, 10, 11, 12, 13, 14, 15);
    #pragma unroll
    for (int ti = 0; ti < 4; ++ti) {
      const int ct = wave * 4 + ti;       // column tile index (0..31)
      // B fragment (CDNA5 16-bit B 32x16 layout): lane holds col N = l16,
      // K chunks [k0+kb8, +8) and [k0+16+kb8, +8), kb8 = lhalf*8. ce16 row-major [C,F] == B^T.
      const _Float16* brow =
          ce16 + ((size_t)(ct * 16 + l16)) * F_DIM + k0 + lhalf * 8;
      v8h b0 = *(const v8h*)(brow);
      v8h b1 = *(const v8h*)(brow + 16);
      v16h b = __builtin_shufflevector(b0, b1, 0, 1, 2, 3, 4, 5, 6, 7,
                                       8, 9, 10, 11, 12, 13, 14, 15);
      acc[ti] = __builtin_amdgcn_wmma_f32_16x16x32_f16(
          /*neg_a=*/false, a, /*neg_b=*/false, b,
          /*c_mod=*/(short)0, acc[ti], /*reuse_a=*/false, /*reuse_b=*/false);
    }
  }

  // Spill scaled logits to LDS. C/D layout: VGPR r -> M = r + lhalf*8, N = l16.
  #pragma unroll
  for (int ti = 0; ti < 4; ++ti) {
    const int cb = wave * 64 + ti * 16 + l16;
    #pragma unroll
    for (int r = 0; r < 8; ++r)
      lds[lhalf * 8 + r][cb] = acc[ti][r] * scale;
  }
  __syncthreads();

  // Fused log_softmax: wave w handles rows 2w and 2w+1; 32 lanes stride cols.
  #pragma unroll
  for (int rr = 0; rr < 2; ++rr) {
    const int row = wave * 2 + rr;
    float mx = -INFINITY;
    for (int c = lane; c < C_DIM; c += 32) mx = fmaxf(mx, lds[row][c]);
    #pragma unroll
    for (int off = 16; off > 0; off >>= 1)
      mx = fmaxf(mx, __shfl_xor(mx, off, 32));
    float se = 0.f;
    for (int c = lane; c < C_DIM; c += 32) se += expf(lds[row][c] - mx);
    #pragma unroll
    for (int off = 16; off > 0; off >>= 1) se += __shfl_xor(se, off, 32);
    const float logZ = mx + logf(se);
    float* out = scores + ((size_t)(pt * 16 + row)) * C_DIM;
    for (int c = lane; c < C_DIM; c += 32) out[c] = lds[row][c] - logZ;
  }
}

// ---------------------------------------------------------------------------
// Kernel 4: segment-mean pooling. One block (256 threads) per caption.
// caption_idx is sorted, so [start,end) found by binary search.
// mapped == -1 wraps to scores row P-1 (matches JAX negative-index gather)
// but still counts as invalid for real_n. Row index is uniform per block ->
// readfirstlane makes the base address scalar (saddr loads); each thread
// streams one float2 (b64) of the 512-col row.
// ---------------------------------------------------------------------------
__device__ __forceinline__ int lower_bound_i(const int* __restrict__ a,
                                             int n, int key) {
  int lo = 0, hi = n;
  while (lo < hi) {
    int mid = (lo + hi) >> 1;
    if (a[mid] < key) lo = mid + 1; else hi = mid;
  }
  return lo;
}

__global__ __launch_bounds__(256) void k_pool(
    const float* __restrict__ scores, const int* __restrict__ c2p,
    const int* __restrict__ p2o, const int* __restrict__ cap_idx,
    float* __restrict__ pooled, float* __restrict__ real_n,
    float* __restrict__ has_pts, int M, int P) {
  const int s = blockIdx.x;
  const int t = threadIdx.x;
  const int start = lower_bound_i(cap_idx, M, s);
  const int end   = lower_bound_i(cap_idx, M, s + 1);

  float a0 = 0.f, a1 = 0.f;
  int ninv = 0;
  for (int m = start; m < end; ++m) {
    const int mapped = p2o[c2p[m]];
    int row = (mapped < 0) ? (P - 1) : mapped;  // -1 wraps to last row
    ninv += (mapped < 0);
    row = __builtin_amdgcn_readfirstlane(row);  // uniform -> scalar base
    const float2 v = *(const float2*)(scores + (size_t)row * C_DIM + 2 * t);
    a0 += v.x;
    a1 += v.y;
  }
  const float rn = (float)((end - start) - ninv);
  const float denom = (rn > 0.f) ? (1.0f / rn) : 0.0f;
  float2 outv = make_float2(a0 * denom, a1 * denom);
  *(float2*)(pooled + (size_t)s * C_DIM + 2 * t) = outv;
  if (t == 0) {
    real_n[s]  = rn;
    has_pts[s] = (rn > 0.f) ? 1.0f : 0.0f;
  }
}

// ---------------------------------------------------------------------------
// Launch.  Inputs (setup_inputs order):
//   0 feats[V*F] f32, 1 caption_embed[C*F] f32, 2 logit_scale_log[1] f32,
//   3 v2p_map[P] i32, 4 c2p_flat[M] i32, 5 p2o[PC] i32,
//   6 caption_idx[M] i32, 7 num_captions (scalar)
// Output: pooled [NCAP*C] ++ real_n [NCAP] ++ has_pts [NCAP]  (f32)
// ---------------------------------------------------------------------------
extern "C" void kernel_launch(void* const* d_in, const int* in_sizes, int n_in,
                              void* d_out, int out_size, void* d_ws,
                              size_t ws_size, hipStream_t stream) {
  const float* feats = (const float*)d_in[0];
  const float* ce    = (const float*)d_in[1];
  const float* lsl   = (const float*)d_in[2];
  const int*   v2p   = (const int*)d_in[3];
  const int*   c2p   = (const int*)d_in[4];
  const int*   p2o   = (const int*)d_in[5];
  const int*   cidx  = (const int*)d_in[6];

  const int P = in_sizes[3];   // 100000 (multiple of 16)
  const int M = in_sizes[4];   // 200000

  // Workspace layout: [ce f16 | af f16 | scores f32]
  char* ws = (char*)d_ws;
  const size_t ce_bytes = (size_t)C_DIM * F_DIM * sizeof(_Float16);  // 512 KB
  const size_t af_bytes = (size_t)P * F_DIM * sizeof(_Float16);      // ~102 MB
  _Float16* ce16   = (_Float16*)ws;
  _Float16* af16   = (_Float16*)(ws + ce_bytes);
  float*    scores = (float*)(ws + ce_bytes + af_bytes);             // ~205 MB

  float* pooled  = (float*)d_out;
  float* real_n  = pooled + (size_t)NCAP_N * C_DIM;
  float* has_pts = real_n + NCAP_N;

  // 1) convert caption embeddings to f16
  k_ce_to_f16<<<(C_DIM * F_DIM + 255) / 256, 256, 0, stream>>>(
      ce, ce16, C_DIM * F_DIM);
  // 2) gather + normalize + convert features
  k_gather_norm<<<P, 256, 0, stream>>>(feats, v2p, af16, P);
  // 3) WMMA GEMM + log_softmax (P/16 row tiles)
  k_gemm_logsoftmax<<<P / 16, 256, 0, stream>>>(af16, ce16, lsl, scores, P);
  // 4) segment-mean pooling per caption
  k_pool<<<NCAP_N, 256, 0, stream>>>(scores, c2p, p2o, cidx, pooled, real_n,
                                     has_pts, M, P);
}
